// Disentangler_51384988729359
// MI455X (gfx1250) — compile-verified, matrix operand order checked by reading, not run on hardware
//
#include <hip/hip_runtime.h>
#include <hip/hip_bf16.h>
#include <math.h>

// ---------------- problem constants (from reference) ----------------
#define TT     16          // T
#define NTOK   65536
#define EE     32          // E
#define NN_    50000
#define NT_    40000
#define CC     2
#define KK     16
#define MM     1024        // M
#define ROWS   512         // T*E : GEMM is (512 x NNP) * (NNP x 1024)
#define NNP    50048       // NN padded to multiple of 64 (K panel size)
#define KSTEPS (NNP / 32)  // 1564
#define NCHUNK (NNP / 64)  // 782 K-panels of 64
#define KSPLIT 8
#define CPS    ((NCHUNK + KSPLIT - 1) / KSPLIT)  // 98 panels per split

// LDS staging: panel row = 64 elems * 2B = 128B + 16B TDM pad = 144B stride
#define ROWB    144
#define ATILE_B (128 * ROWB)          // 18432 B
#define STAGE_B (2 * ATILE_B)         // A + B panel = 36864 B
#define SMEM_B  (2 * STAGE_B)         // double buffered = 73728 B

typedef __attribute__((ext_vector_type(8)))  float        v8f;
typedef __attribute__((ext_vector_type(16))) __bf16       v16bf;
typedef __attribute__((ext_vector_type(4)))  unsigned int v4u;
typedef __attribute__((ext_vector_type(4)))  int          v4i;
typedef __attribute__((ext_vector_type(8)))  int          v8i;

union Frag16 { uint4 q[2]; v16bf v; };

__device__ __forceinline__ unsigned short f32_to_bf16(float f) {
    unsigned int u = __float_as_uint(f);
    unsigned int lsb = (u >> 16) & 1u;
    u += 0x7fffu + lsb;                 // round to nearest even
    return (unsigned short)(u >> 16);
}

__device__ __forceinline__ float gelu_exact(float x) {
    return 0.5f * x * (1.0f + erff(x * 0.70710678118654752f));
}

// ---------------- TDM: load a 2D tile (128 rows x 64 elems of bf16) into LDS ----------------
// D# group0: count=1 | lds_addr | global_addr | type=2
// D# group1: data_size=2B, pad_enable, pad_interval=32 DW, pad_amount=4 DW,
//            tensor_dim0=NNP, tensor_dim1=rows, tile_dim0=64, tile_dim1=128,
//            tensor_dim0_stride=NNP
__device__ __forceinline__ void tdm_load_tile(const unsigned short* gptr,
                                              unsigned ldsOff, unsigned tensorRows) {
    unsigned long long ga = (unsigned long long)(size_t)gptr;
    v4u g0;
    g0[0] = 1u;                                   // count=1, user descriptor
    g0[1] = ldsOff;                               // lds_addr (byte offset)
    g0[2] = (unsigned)ga;                         // global_addr[31:0]
    g0[3] = (unsigned)((ga >> 32) & 0x01ffffffu)  // global_addr[56:32]
          | (2u << 30);                           // type = 2 ("image")
    v8i g1;
    g1[0] = (int)((1u << 16)      // data_size = 2 bytes
                | (1u << 20)      // pad_enable
                | (4u << 22)      // pad_interval: 32 DWORDs (=128B row)
                | (3u << 25));    // pad_amount: 4 DWORDs (=16B)
    g1[1] = (int)(((unsigned)NNP & 0xffffu) << 16);                    // tensor_dim0[15:0]
    g1[2] = (int)((((unsigned)NNP >> 16) & 0xffffu)
                | ((tensorRows & 0xffffu) << 16));                     // d0[31:16] | d1[15:0]
    g1[3] = (int)(((tensorRows >> 16) & 0xffffu) | (64u << 16));       // d1[31:16] | tile_dim0=64
    g1[4] = (int)(128u);                                               // tile_dim1=128, tile_dim2=0
    g1[5] = (int)((unsigned)NNP);                                      // tensor_dim0_stride lo
    g1[6] = 0;
    g1[7] = 0;
    v4i z4 = {0, 0, 0, 0};
#if defined(__clang_major__) && (__clang_major__ >= 23)
    v8i z8 = {0, 0, 0, 0, 0, 0, 0, 0};
    __builtin_amdgcn_tensor_load_to_lds(g0, g1, z4, z4, z8, 0);
#else
    __builtin_amdgcn_tensor_load_to_lds(g0, g1, z4, z4, 0);
#endif
}

// ---------------- kernel 0: zero f32 accumulators ----------------
__global__ void k_zero(float4* __restrict__ p, int n4) {
    int i = blockIdx.x * blockDim.x + threadIdx.x;
    if (i < n4) p[i] = make_float4(0.f, 0.f, 0.f, 0.f);
}

// ---------------- kernel 1: LayerNorm + scatter-add -> A_f32[t][e][n] ----------------
__global__ void k_ln_scatter(const float* __restrict__ x,
                             const float* __restrict__ ln_w,
                             const float* __restrict__ ln_b,
                             const int*   __restrict__ idx,
                             float*       __restrict__ Af32) {
    int job = blockIdx.x * blockDim.x + threadIdx.x;
    if (job >= TT * NT_) return;
    int t = job / NT_, tok = job % NT_;
    const float* xp = x + ((size_t)t * NTOK + tok) * EE;

    float v[EE];
    const float4* x4 = (const float4*)xp;
    #pragma unroll
    for (int j = 0; j < EE / 4; ++j) {
        float4 f = x4[j];
        v[4*j+0] = f.x; v[4*j+1] = f.y; v[4*j+2] = f.z; v[4*j+3] = f.w;
    }
    float mu = 0.f;
    #pragma unroll
    for (int e = 0; e < EE; ++e) mu += v[e];
    mu *= (1.0f / EE);
    float var = 0.f;
    #pragma unroll
    for (int e = 0; e < EE; ++e) { float d = v[e] - mu; var += d * d; }
    var *= (1.0f / EE);
    float rinv = rsqrtf(var + 1e-5f);

    int n = idx[t * NT_ + tok];
    float* base = Af32 + (size_t)t * EE * NN_ + n;
    #pragma unroll
    for (int e = 0; e < EE; ++e) {
        float val = (v[e] - mu) * rinv * ln_w[e] + ln_b[e];
        atomicAdd(base + (size_t)e * NN_, val);
    }
}

// ---------------- kernel 2: A f32 -> bf16, pad K to NNP ----------------
__global__ void k_convA(const float* __restrict__ Af32, unsigned short* __restrict__ Abf) {
    size_t i = (size_t)blockIdx.x * blockDim.x + threadIdx.x;
    const size_t tot = (size_t)ROWS * NNP;
    if (i >= tot) return;
    int np = (int)(i % NNP);
    size_t row = i / NNP;
    float f = (np < NN_) ? Af32[row * NN_ + np] : 0.f;
    Abf[i] = f32_to_bf16(f);
}

// ---------------- kernel 3: Bt[m][n] = bf16(mlp_w[n][m]), LDS-tiled transpose ----------------
__global__ void k_transB(const float* __restrict__ mlp_w, unsigned short* __restrict__ Bt) {
    __shared__ float tile[32][33];
    int n0 = blockIdx.x * 32;   // 1564 blocks -> covers NNP exactly
    int m0 = blockIdx.y * 32;   // 32 blocks
    int tx = threadIdx.x, ty = threadIdx.y;        // (32, 8)
    #pragma unroll
    for (int i = 0; i < 4; ++i) {
        int n = n0 + ty + i * 8;
        tile[ty + i * 8][tx] = (n < NN_) ? mlp_w[(size_t)n * MM + (m0 + tx)] : 0.f;
    }
    __syncthreads();
    #pragma unroll
    for (int i = 0; i < 4; ++i) {
        int m = m0 + ty + i * 8;
        Bt[(size_t)m * NNP + n0 + tx] = f32_to_bf16(tile[tx][ty + i * 8]);
    }
}

// ---------------- kernel 4: TDM double-buffered WMMA GEMM ----------------
// grid: x=8 (col blocks of 128), y=4 (row blocks of 128), z=KSPLIT; block=256 (8 waves)
// wave (wr=0..3, wc=0..1) computes 32 rows x 64 cols: 2 A-frags x 4 B-frags -> 8 wmma/k-step.
// Wave 0 issues 2 TDM tile loads per stage (EXEC ignored by TDM), double buffered.
__global__ void __launch_bounds__(256)
k_gemm(const unsigned short* __restrict__ A,
       const unsigned short* __restrict__ Bt,
       float* __restrict__ Cacc) {
    extern __shared__ char smem[];
    const int wv   = threadIdx.x >> 5;
    const int lane = threadIdx.x & 31;
    const int wr   = wv & 3;          // row group within block (4 x 32 rows)
    const int wc   = wv >> 2;         // col group within block (2 x 64 cols)
    const int col  = lane & 15;
    const int half = lane >> 4;

    const int rowBase = blockIdx.y * 128;
    const int colBase = blockIdx.x * 128;
    int c0 = blockIdx.z * CPS;
    int c1 = c0 + CPS; if (c1 > NCHUNK) c1 = NCHUNK;

    const unsigned smemOff = (unsigned)(size_t)(&smem[0]);   // LDS byte offset

    // prologue: stage c0 into buffer 0
    if (wv == 0) {
        tdm_load_tile(A  + (size_t)rowBase * NNP + (size_t)c0 * 64, smemOff,           ROWS);
        tdm_load_tile(Bt + (size_t)colBase * NNP + (size_t)c0 * 64, smemOff + ATILE_B, MM);
    }

    v8f acc[2][4];
    #pragma unroll
    for (int i = 0; i < 2; ++i)
        #pragma unroll
        for (int j = 0; j < 4; ++j) acc[i][j] = v8f{};

    for (int c = c0; c < c1; ++c) {
        int b = (c - c0) & 1;
        if (wv == 0) {
            if (c + 1 < c1) {
                unsigned nb = smemOff + (unsigned)((b ^ 1) * STAGE_B);
                tdm_load_tile(A  + (size_t)rowBase * NNP + (size_t)(c + 1) * 64, nb,           ROWS);
                tdm_load_tile(Bt + (size_t)colBase * NNP + (size_t)(c + 1) * 64, nb + ATILE_B, MM);
                __builtin_amdgcn_s_wait_tensorcnt(2);  // stage c done, c+1 in flight
            } else {
                __builtin_amdgcn_s_wait_tensorcnt(0);
            }
        }
        __syncthreads();   // publish stage c panels to all waves

        const char* Abuf = smem + b * STAGE_B;
        const char* Bbuf = Abuf + ATILE_B;
        #pragma unroll
        for (int ks = 0; ks < 2; ++ks) {
            int kb = ks * 64 + half * 16;   // byte offset of fragment elements 0..7
            Frag16 af[2], bf[4];
            #pragma unroll
            for (int i = 0; i < 2; ++i) {
                const char* p = Abuf + (size_t)(wr * 32 + i * 16 + col) * ROWB + kb;
                af[i].q[0] = *(const uint4*)(p);
                af[i].q[1] = *(const uint4*)(p + 32);
            }
            #pragma unroll
            for (int j = 0; j < 4; ++j) {
                const char* p = Bbuf + (size_t)(wc * 64 + j * 16 + col) * ROWB + kb;
                bf[j].q[0] = *(const uint4*)(p);
                bf[j].q[1] = *(const uint4*)(p + 32);
            }
            #pragma unroll
            for (int i = 0; i < 2; ++i)
                #pragma unroll
                for (int j = 0; j < 4; ++j)
                    acc[i][j] = __builtin_amdgcn_wmma_f32_16x16x32_bf16(
                        false, af[i].v, false, bf[j].v, (short)0, acc[i][j], false, false);
        }
        __syncthreads();   // all waves done reading stage c; its buffer may be overwritten
    }

    // C/D layout: lane L, vgpr r -> M = r + 8*(L/16), N = L%16
    #pragma unroll
    for (int i = 0; i < 2; ++i) {
        #pragma unroll
        for (int j = 0; j < 4; ++j) {
            float* cb = Cacc + (size_t)(rowBase + wr * 32 + i * 16 + half * 8) * MM
                             + (colBase + wc * 64 + j * 16 + col);
            #pragma unroll
            for (int r = 0; r < 8; ++r)
                atomicAdd(cb + (size_t)r * MM, acc[i][j][r]);
        }
    }
}

// ---------------- kernel 5: bias + GELU + stride-16 conv + GELU -> out[t][c*64+w] ----------------
__global__ void k_epilogue(const float* __restrict__ Cacc,
                           const float* __restrict__ mlp_b,
                           const float* __restrict__ conv_w,
                           const float* __restrict__ conv_b,
                           float* __restrict__ out) {
    int wave = threadIdx.x >> 5, lane = threadIdx.x & 31;
    int job = blockIdx.x * 4 + wave;
    if (job >= TT * 64) return;            // whole-wave uniform branch
    int t = job / 64, w = job % 64;
    int i = lane >> 4, kh = lane & 15;

    const float* crow = Cacc + ((size_t)t * EE + lane) * MM + w * 16;
    const float* bb = mlp_b + w * 16;
    float s0 = 0.f, s1 = 0.f;
    #pragma unroll
    for (int kw = 0; kw < 16; ++kw) {
        float a = gelu_exact(crow[kw] + bb[kw]);
        s0 += a * conv_w[((0 * CC + i) * KK + kh) * KK + kw];
        s1 += a * conv_w[((1 * CC + i) * KK + kh) * KK + kw];
    }
    #pragma unroll
    for (int off = 16; off >= 1; off >>= 1) {
        s0 += __shfl_xor(s0, off, 32);
        s1 += __shfl_xor(s1, off, 32);
    }
    if (lane == 0) {
        out[t * 128 +      w] = gelu_exact(s0 + conv_b[0]);
        out[t * 128 + 64 + w] = gelu_exact(s1 + conv_b[1]);
    }
}

// ---------------- launch ----------------
extern "C" void kernel_launch(void* const* d_in, const int* in_sizes, int n_in,
                              void* d_out, int out_size, void* d_ws, size_t ws_size,
                              hipStream_t stream) {
    const float* x      = (const float*)d_in[0];
    const float* ln_w   = (const float*)d_in[1];
    const float* ln_b   = (const float*)d_in[2];
    const float* mlp_w  = (const float*)d_in[3];
    const float* mlp_b  = (const float*)d_in[4];
    const float* conv_w = (const float*)d_in[5];
    const float* conv_b = (const float*)d_in[6];
    const int*   idx    = (const int*)d_in[7];
    float* out = (float*)d_out;

    // workspace layout (bytes, 256-aligned)
    const size_t szAf32 = (size_t)TT * EE * NN_ * 4;         // 102,400,000
    const size_t szCacc = (size_t)ROWS * MM * 4;             //   2,097,152
    const size_t szAbf  = (size_t)ROWS * NNP * 2;            //  51,249,152
    char* ws = (char*)d_ws;
    float*          Af32 = (float*)(ws);
    float*          Cacc = (float*)(ws + szAf32);
    unsigned short* Abf  = (unsigned short*)(ws + szAf32 + szCacc);
    unsigned short* Bt   = (unsigned short*)(ws + szAf32 + szCacc + szAbf);

    // 0) zero the two f32 accumulators (contiguous)
    {
        int n4 = (int)((szAf32 + szCacc) / 16);
        k_zero<<<(n4 + 255) / 256, 256, 0, stream>>>((float4*)ws, n4);
    }
    // 1) layer-norm + scatter
    {
        int jobs = TT * NT_;
        k_ln_scatter<<<(jobs + 255) / 256, 256, 0, stream>>>(x, ln_w, ln_b, idx, Af32);
    }
    // 2) A -> bf16 (padded)
    {
        size_t tot = (size_t)ROWS * NNP;
        k_convA<<<(int)((tot + 255) / 256), 256, 0, stream>>>(Af32, Abf);
    }
    // 3) mlp_w transpose -> bf16
    {
        dim3 g(NNP / 32, MM / 32), b(32, 8);
        k_transB<<<g, b, 0, stream>>>(mlp_w, Bt);
    }
    // 4) TDM + WMMA GEMM (split-K, atomic f32 accumulate)
    {
        dim3 g(MM / 128, ROWS / 128, KSPLIT);
        k_gemm<<<g, 256, SMEM_B, stream>>>(Abf, Bt, Cacc);
    }
    // 5) epilogue: bias+gelu+conv+gelu
    {
        int jobs = TT * 64;
        k_epilogue<<<(jobs + 3) / 4, 128, 0, stream>>>(Cacc, mlp_b, conv_w, conv_b, out);
    }
    (void)in_sizes; (void)n_in; (void)out_size; (void)ws_size;
}
